// HHNodeMP_42683384987780
// MI455X (gfx1250) — matrix operand order repaired and checked by reading, not compiled, not run on hardware
//
#include <hip/hip_runtime.h>
#include <hip/hip_bf16.h>

typedef __bf16 bf16;
typedef __attribute__((ext_vector_type(16))) __bf16 v16bf;
typedef __attribute__((ext_vector_type(8)))  __bf16 v8bf;
typedef __attribute__((ext_vector_type(4)))  __bf16 v4bf;
typedef __attribute__((ext_vector_type(8)))  float  v8f;

#define DIM 256
#define NKT 8      // 256 / K32
#define NJT 16     // 256 / 16
#define WAVES 4
#define WAVE_LDS 40960   // cur/val 8K + spa 8K + tmp 8K + feaS 8K + feaT 8K

// ---------------------------------------------------------------------------
// Prep: convert the 4 weight matrices f32 -> bf16, pre-swizzled into per-lane
// WMMA B-fragments:  wfrag[mat][kt][jt][lane][16 bf16]  (32B contiguous/lane).
// B order used here (contiguous K, same as A): lane<16 -> col=lane, K=e ; lane>=16 -> col=lane-16,
// K=16+e.  mat 0..2: B[k][j]=W[k][j].  mat 3 (theta, x@W.T): B[k][j]=W[j][k].
// ---------------------------------------------------------------------------
__global__ void hh_prep_weights(const float* __restrict__ nodeW,
                                const float* __restrict__ spaW,
                                const float* __restrict__ tmpW,
                                const float* __restrict__ thetaW,
                                bf16* __restrict__ wfrag) {
  int idx  = blockIdx.x * blockDim.x + threadIdx.x;   // 0 .. 262143
  int e    = idx & 15;
  int lane = (idx >> 4) & 31;
  int jt   = (idx >> 9) & 15;
  int kt   = (idx >> 13) & 7;
  int mat  = (idx >> 16) & 3;
  int kg = kt * 32 + ((lane >> 4) << 4) + e;
  int jg = jt * 16 + (lane & 15);
  const float* W = (mat == 0) ? nodeW : (mat == 1) ? spaW : (mat == 2) ? tmpW : thetaW;
  float v = (mat < 3) ? W[kg * DIM + jg] : W[jg * DIM + kg];
  wfrag[idx] = (bf16)v;
}

// ---------------------------------------------------------------------------
// Main: one wave per 16-row tile of N.
// ---------------------------------------------------------------------------
__global__ __launch_bounds__(WAVES * 32)
void hh_main(const float* __restrict__ cur,
             const float* __restrict__ spaE,
             const float* __restrict__ tmpE,
             const bf16*  __restrict__ wfrag,
             const float* __restrict__ thetaB,
             float* __restrict__ out,
             int n_tiles, int Ntot) {
  extern __shared__ char smem[];
  const int lane = threadIdx.x & 31;
  const int wave = threadIdx.x >> 5;
  const int tile = blockIdx.x * WAVES + wave;
  if (tile >= n_tiles) return;              // wave-uniform; no barriers used
  const int row_base = tile * 16;

  char* lds    = smem + wave * WAVE_LDS;
  bf16* ldsCur = (bf16*)(lds);              // reused for `val` in stage 3
  bf16* ldsSpa = (bf16*)(lds + 8192);
  bf16* ldsTmp = (bf16*)(lds + 16384);
  bf16* feaS   = (bf16*)(lds + 24576);
  bf16* feaT   = (bf16*)(lds + 32768);
  float* attnS = (float*)(lds + 8192);      // spa staging dead after stage 1
  float* attnT = (float*)(lds + 8192 + 64);

  // ---- stage 16 rows x 256 cols of each input into LDS as bf16 ----
  auto stage = [&](const float* src, bf16* dst) {
    const float4* s4 = (const float4*)(src + (size_t)row_base * DIM);
    for (int i = lane; i < 16 * (DIM / 4); i += 32) {
      float4 v = make_float4(0.f, 0.f, 0.f, 0.f);
      if (row_base + (i >> 6) < Ntot) v = s4[i];
      v4bf p;
      p[0] = (bf16)v.x; p[1] = (bf16)v.y; p[2] = (bf16)v.z; p[3] = (bf16)v.w;
      *(v4bf*)(dst + i * 4) = p;
    }
  };
  stage(cur, ldsCur);
  stage(spaE, ldsSpa);
  stage(tmpE, ldsTmp);
  __asm volatile("s_wait_dscnt 0" ::: "memory");

  // A fragment (16x32 bf16) from LDS: lane m=lane&15, K = 16*(lane>=16) + e
  auto loadA = [&](const bf16* base, int kt) -> v16bf {
    const bf16* p = base + (lane & 15) * DIM + kt * 32 + (lane >> 4) * 16;
    v8bf lo = *(const v8bf*)(p);        // cols +0..7
    v8bf hi = *(const v8bf*)(p + 8);    // cols +8..15
    return __builtin_shufflevector(lo, hi, 0,1,2,3,4,5,6,7,8,9,10,11,12,13,14,15);
  };
  auto loadB = [&](int mat, int kt, int jt) -> v16bf {
    return *(const v16bf*)(wfrag + ((((mat * NKT + kt) * NJT + jt) * 32 + lane) << 4));
  };

  // ---- stage 1: node/spa/tmp GEMMs + in-register attention dots ----
  float dS[8], dT[8];
  for (int r = 0; r < 8; ++r) { dS[r] = 0.f; dT[r] = 0.f; }
  const int colL = lane & 15;
  const int mb   = (lane >> 4) * 8;

  for (int jt = 0; jt < NJT; ++jt) {
    v8f aN = {}, aS = {}, aT = {};
#pragma unroll
    for (int kt = 0; kt < NKT; ++kt) {
      v16bf ac = loadA(ldsCur, kt);
      v16bf as = loadA(ldsSpa, kt);
      v16bf at = loadA(ldsTmp, kt);
      aN = __builtin_amdgcn_wmma_f32_16x16x32_bf16(false, ac, false, loadB(0, kt, jt), (short)0, aN, false, false);
      aS = __builtin_amdgcn_wmma_f32_16x16x32_bf16(false, as, false, loadB(1, kt, jt), (short)0, aS, false, false);
      aT = __builtin_amdgcn_wmma_f32_16x16x32_bf16(false, at, false, loadB(2, kt, jt), (short)0, aT, false, false);
    }
    int col = jt * 16 + colL;
#pragma unroll
    for (int r = 0; r < 8; ++r) {
      dS[r] += aS[r] * aN[r];           // same C layout -> elementwise dot
      dT[r] += aT[r] * aN[r];
      feaS[(mb + r) * DIM + col] = (bf16)aS[r];
      feaT[(mb + r) * DIM + col] = (bf16)aT[r];
    }
  }
  // reduce dots over the 16-lane column groups
#pragma unroll
  for (int m = 1; m <= 8; m <<= 1)
#pragma unroll
    for (int r = 0; r < 8; ++r) {
      dS[r] += __shfl_xor(dS[r], m, 32);
      dT[r] += __shfl_xor(dT[r], m, 32);
    }
  __asm volatile("s_wait_dscnt 0" ::: "memory");
  if ((lane & 15) == 0) {               // lane 0 -> rows 0..7, lane 16 -> rows 8..15
#pragma unroll
    for (int r = 0; r < 8; ++r) {
      attnS[mb + r] = dS[r] * 0.0625f;  // 1/sqrt(256)
      attnT[mb + r] = dT[r] * 0.0625f;
    }
  }
  __asm volatile("s_wait_dscnt 0" ::: "memory");

  // ---- stage 2: val = spa_attn*spa_fea + tmp_attn*tmp_fea (bf16, into cur slot)
  for (int i = lane; i < 16 * DIM; i += 32) {
    int m = i >> 8;
    float v = attnS[m] * (float)feaS[i] + attnT[m] * (float)feaT[i];
    ldsCur[i] = (bf16)v;
  }
  __asm volatile("s_wait_dscnt 0" ::: "memory");

  // ---- stage 3: out = relu(val @ theta.T + b) ----
  for (int jt = 0; jt < NJT; ++jt) {
    v8f acc = {};
#pragma unroll
    for (int kt = 0; kt < NKT; ++kt)
      acc = __builtin_amdgcn_wmma_f32_16x16x32_bf16(false, loadA(ldsCur, kt), false, loadB(3, kt, jt), (short)0, acc, false, false);
    int col = jt * 16 + colL;
    float bias = thetaB[col];
    float* o = out + (size_t)(row_base + mb) * DIM + col;
#pragma unroll
    for (int r = 0; r < 8; ++r) {
      if (row_base + mb + r < Ntot) {
        float v = acc[r] + bias;
        o[(size_t)r * DIM] = v > 0.f ? v : 0.f;
      }
    }
  }
}

// ---------------------------------------------------------------------------
extern "C" void kernel_launch(void* const* d_in, const int* in_sizes, int n_in,
                              void* d_out, int out_size, void* d_ws, size_t ws_size,
                              hipStream_t stream) {
  const float* cur    = (const float*)d_in[0];
  const float* spaE   = (const float*)d_in[1];
  const float* tmpE   = (const float*)d_in[2];
  const float* nodeW  = (const float*)d_in[3];
  const float* spaW   = (const float*)d_in[4];
  const float* tmpW   = (const float*)d_in[5];
  const float* thetaW = (const float*)d_in[6];
  const float* thetaB = (const float*)d_in[7];
  float* out  = (float*)d_out;
  bf16* wfrag = (bf16*)d_ws;          // needs 4*8*16*32*16*2 = 512 KB

  int Ntot = in_sizes[0] / DIM;
  // swizzle weights into WMMA B-fragments (262144 bf16 elements)
  hh_prep_weights<<<1024, 256, 0, stream>>>(nodeW, spaW, tmpW, thetaW, wfrag);

  int n_tiles = (Ntot + 15) / 16;
  int blocks  = (n_tiles + WAVES - 1) / WAVES;
  hh_main<<<blocks, WAVES * 32, WAVES * WAVE_LDS, stream>>>(
      cur, spaE, tmpE, wfrag, thetaB, out, n_tiles, Ntot);
}
